// TensorBase_60679297958560
// MI455X (gfx1250) — matrix operand order repaired and checked by reading, not compiled, not run on hardware
//
#include <hip/hip_runtime.h>
#include <hip/hip_bf16.h>

// ---------------------------------------------------------------------------
// TensoRF-style NeRF renderer for MI455X (gfx1250, wave32, WMMA bf16).
// Pipeline: prep(weights->bf16 swizzled) -> points(sigma+feat) ->
//           composite(weights/depth) ->
//           MLP (TDM tensor_load_to_lds weight staging + v_wmma_f32_16x16x32_bf16)
//           -> final reduce.
// ---------------------------------------------------------------------------

#define N_RAYS   1024
#define N_SAMP   512
#define NPTS     (N_RAYS * N_SAMP)     // 524288
#define GRIDN    300
#define PLANE_HW (GRIDN * GRIDN)       // 90000
#define DN       16
#define AN       16
#define APP_DIM  27
#define FEAT_C   128
#define IN_MLP   150                   // padded to 160 for K
#define KPAD     160
#define TILE_M   128                   // points per MLP workgroup
#define STEP_F   ((3.0f / 299.0f) * 0.5f)
#define DIST_SCALE 25.0f
#define DENS_SHIFT (-10.0f)
#define W_THRES  1e-4f

#define W1_DW    (5 * 128 * 32 / 2)    // 10240 dwords (bf16 pairs)
#define W2_DW    (4 * 128 * 32 / 2)    // 8192 dwords
#define W3_DW    (4 * 16 * 32 / 2)     // 1024 dwords

typedef __attribute__((ext_vector_type(16))) __bf16        v16bf;
typedef __attribute__((ext_vector_type(8)))  float         v8f;
typedef __attribute__((ext_vector_type(4))) unsigned int  u32x4;
typedef __attribute__((ext_vector_type(8)))  int           i32x8;
typedef __attribute__((ext_vector_type(4)))  int           i32x4;

union Frag { v16bf v; u32x4 q[2]; };

// ---- CDNA5 Tensor Data Mover availability (device pass only) --------------
#if defined(__HIP_DEVICE_COMPILE__)
#  if __has_builtin(__builtin_amdgcn_tensor_load_to_lds) && \
      __has_builtin(__builtin_amdgcn_s_wait_tensorcnt)
#    define HAVE_TDM 1
#  endif
#endif

#if defined(HAVE_TDM)
// 1-D dword copy global -> LDS through the TDM (D# per CDNA5 ISA ch.8):
// group0: count=1, lds_addr, global_addr[56:0], type=2
// group1: data_size=4B, tensor_dim0 = ndw, tensor_dim1 = 1,
//         tile_dim0 = ndw (<=65535), tile_dim1/2 = 0 (unused),
//         tensor_dim0_stride = ndw.  groups 2/3 zero (<=2-D tensor).
__device__ __forceinline__ void tdm_load_dwords(const void* lds_dst,
                                                const void* gsrc,
                                                unsigned ndw) {
  unsigned long long ga = (unsigned long long)gsrc;
  unsigned lds = (unsigned)(unsigned long long)lds_dst; // LDS byte offset = addr[31:0]
  u32x4 g0;
  g0[0] = 1u;                                              // count=1
  g0[1] = lds;                                             // lds_addr
  g0[2] = (unsigned)ga;                                    // global_addr[31:0]
  g0[3] = (unsigned)((ga >> 32) & 0x01FFFFFFull) | 0x80000000u; // [56:32] + type=2
  i32x8 g1;
  g1[0] = (2 << 16);                       // data_size = 4 bytes
  g1[1] = (int)((ndw & 0xFFFFu) << 16);    // tensor_dim0[15:0]
  g1[2] = (int)(((ndw >> 16) & 0xFFFFu) | (1u << 16)); // tensor_dim0[31:16], tensor_dim1=1
  g1[3] = (int)((ndw & 0xFFFFu) << 16);    // tile_dim0 = ndw
  g1[4] = 0;                               // tile_dim1 = 0, tile_dim2 = 0
  g1[5] = (int)ndw;                        // tensor_dim0_stride[31:0]
  g1[6] = 0;
  g1[7] = 0;
  i32x4 z4 = (i32x4)0;
#if __clang_major__ >= 23
  i32x8 z8 = (i32x8)0;
  __builtin_amdgcn_tensor_load_to_lds(g0, g1, z4, z4, z8, 0);
#else
  __builtin_amdgcn_tensor_load_to_lds(g0, g1, z4, z4, 0);
#endif
}
#endif

__device__ __forceinline__ void stage_weights(unsigned short* dst,
                                              const unsigned short* src,
                                              int ndw, int t) {
#if defined(HAVE_TDM)
  if (t < 32) tdm_load_dwords(dst, src, (unsigned)ndw); // one TDM op from wave 0
#else
  const unsigned int* s = (const unsigned int*)src;
  unsigned int* d = (unsigned int*)dst;
  for (int i = t; i < ndw; i += 256) d[i] = s[i];
#endif
}

template <int N>
__device__ __forceinline__ void wait_tensor(int t) {
#if defined(HAVE_TDM)
  if (t < 32) __builtin_amdgcn_s_wait_tensorcnt(N);  // immediate operand required
#else
  (void)t;
#endif
}

__device__ __forceinline__ unsigned short f2bf(float f) {
  unsigned int u = __float_as_uint(f);
  return (unsigned short)((u + 0x7FFFu + ((u >> 16) & 1u)) >> 16);
}

__device__ __forceinline__ void interp_setup(float c, int n, int& i0, float& w) {
  float t = (c * 0.5f + 0.5f) * (float)(n - 1);
  t = fminf(fmaxf(t, 0.0f), (float)(n - 1));
  float f = floorf(t);
  if (f > (float)(n - 2)) f = (float)(n - 2);
  i0 = (int)f;
  w = t - f;
}

// sum over 16 channels of bilinear(plane) * linear(line)
__device__ __forceinline__ float plane_line_dot(const float* __restrict__ pl,
                                                const float* __restrict__ ln,
                                                float x, float y, float z) {
  int x0, y0, z0; float wx, wy, wz;
  interp_setup(x, GRIDN, x0, wx);
  interp_setup(y, GRIDN, y0, wy);
  interp_setup(z, GRIDN, z0, wz);
  int base = y0 * GRIDN + x0;
  float acc = 0.0f;
#pragma unroll
  for (int c = 0; c < DN; ++c) {
    const float* q = pl + c * PLANE_HW + base;
    float v00 = q[0], v01 = q[1], v10 = q[GRIDN], v11 = q[GRIDN + 1];
    float pv = (v00 * (1.0f - wx) + v01 * wx) * (1.0f - wy) +
               (v10 * (1.0f - wx) + v11 * wx) * wy;
    float lv = ln[c * GRIDN + z0] * (1.0f - wz) + ln[c * GRIDN + z0 + 1] * wz;
    acc += pv * lv;
  }
  return acc;
}

__device__ __forceinline__ void plane_line_coef(const float* __restrict__ pl,
                                                const float* __restrict__ ln,
                                                float x, float y, float z,
                                                float* __restrict__ out) {
  int x0, y0, z0; float wx, wy, wz;
  interp_setup(x, GRIDN, x0, wx);
  interp_setup(y, GRIDN, y0, wy);
  interp_setup(z, GRIDN, z0, wz);
  int base = y0 * GRIDN + x0;
#pragma unroll
  for (int c = 0; c < AN; ++c) {
    const float* q = pl + c * PLANE_HW + base;
    float v00 = q[0], v01 = q[1], v10 = q[GRIDN], v11 = q[GRIDN + 1];
    float pv = (v00 * (1.0f - wx) + v01 * wx) * (1.0f - wy) +
               (v10 * (1.0f - wx) + v11 * wx) * wy;
    float lv = ln[c * GRIDN + z0] * (1.0f - wz) + ln[c * GRIDN + z0 + 1] * wz;
    out[c] = pv * lv;
  }
}

__device__ __forceinline__ float ray_tmin(const float* __restrict__ ro,
                                          const float* __restrict__ rd,
                                          int ray, float* o, float* d) {
  float tmin = -1e30f;
#pragma unroll
  for (int i = 0; i < 3; ++i) {
    o[i] = ro[ray * 3 + i];
    d[i] = rd[ray * 3 + i];
    float v  = (d[i] == 0.0f) ? 1e-6f : d[i];
    float ra = ( 1.5f - o[i]) / v;
    float rb = (-1.5f - o[i]) / v;
    tmin = fmaxf(tmin, fminf(ra, rb));
  }
  return fminf(fmaxf(tmin, 2.0f), 6.0f);
}

// --------------------------- kernel 0: weight prep -------------------------
// W1B: [5][128][32] bf16 (K padded 150->160), W2B: [4][128][32], W3B: [4][16][32]
__global__ void k_prep(const float* __restrict__ w1, const float* __restrict__ w2,
                       const float* __restrict__ w3,
                       unsigned short* __restrict__ W1B,
                       unsigned short* __restrict__ W2B,
                       unsigned short* __restrict__ W3B) {
  int i = blockIdx.x * blockDim.x + threadIdx.x;
  if (i < 5 * 128 * 32) {
    int kk = i & 31, n = (i >> 5) & 127, kc = i >> 12;
    int k = kc * 32 + kk;
    W1B[i] = (k < IN_MLP) ? f2bf(w1[n * IN_MLP + k]) : (unsigned short)0;
  } else if (i < 5 * 128 * 32 + 4 * 128 * 32) {
    int j = i - 5 * 128 * 32;
    int kk = j & 31, n = (j >> 5) & 127, kc = j >> 12;
    W2B[j] = f2bf(w2[n * 128 + kc * 32 + kk]);
  } else if (i < 5 * 128 * 32 + 4 * 128 * 32 + 4 * 16 * 32) {
    int j = i - (5 * 128 * 32 + 4 * 128 * 32);
    int kk = j & 31, n = (j >> 5) & 15, kc = j >> 9;
    W3B[j] = (n < 3) ? f2bf(w3[n * 128 + kc * 32 + kk]) : (unsigned short)0;
  }
}

// --------------------------- kernel 1: per-point ---------------------------
__global__ void k_points(const float* __restrict__ ro, const float* __restrict__ rd,
                         const float* __restrict__ dp0, const float* __restrict__ dl0,
                         const float* __restrict__ ap0, const float* __restrict__ al0,
                         const float* __restrict__ dp1, const float* __restrict__ dl1,
                         const float* __restrict__ ap1, const float* __restrict__ al1,
                         const float* __restrict__ dp2, const float* __restrict__ dl2,
                         const float* __restrict__ ap2, const float* __restrict__ al2,
                         const float* __restrict__ basis,
                         float* __restrict__ gsigma, float* __restrict__ gfeat) {
  int p = blockIdx.x * blockDim.x + threadIdx.x;
  if (p >= NPTS) return;
  int ray = p >> 9, s = p & (N_SAMP - 1);
  float o[3], d[3];
  float tmin = ray_tmin(ro, rd, ray, o, d);
  float z = tmin + STEP_F * (float)s;
  float pt[3], xn[3];
  bool valid = true;
#pragma unroll
  for (int i = 0; i < 3; ++i) {
    pt[i] = o[i] + d[i] * z;
    valid = valid && (pt[i] >= -1.5f) && (pt[i] <= 1.5f);
    xn[i] = pt[i] * (1.0f / 1.5f);
  }
  // density: MAT=((0,1),(0,2),(1,2)), VEC=(2,1,0)
  float sf = 0.0f;
  sf += plane_line_dot(dp0, dl0, xn[0], xn[1], xn[2]);
  sf += plane_line_dot(dp1, dl1, xn[0], xn[2], xn[1]);
  sf += plane_line_dot(dp2, dl2, xn[1], xn[2], xn[0]);
  float a = sf + DENS_SHIFT;
  float sp = (a > 20.0f) ? a : log1pf(__expf(a));
  gsigma[p] = valid ? sp : 0.0f;
  // appearance coefficients -> 27-dim feature
  float coef[3 * AN];
  plane_line_coef(ap0, al0, xn[0], xn[1], xn[2], coef);
  plane_line_coef(ap1, al1, xn[0], xn[2], xn[1], coef + AN);
  plane_line_coef(ap2, al2, xn[1], xn[2], xn[0], coef + 2 * AN);
  size_t fo = (size_t)p * APP_DIM;
  for (int aidx = 0; aidx < APP_DIM; ++aidx) {
    float f = 0.0f;
#pragma unroll
    for (int k = 0; k < 3 * AN; ++k) f += coef[k] * basis[aidx * (3 * AN) + k];
    gfeat[fo + aidx] = f;
  }
}

// --------------------------- kernel 2: compositing -------------------------
__global__ void k_composite(const float* __restrict__ ro, const float* __restrict__ rd,
                            const float* __restrict__ gsigma,
                            float* __restrict__ gweight, float* __restrict__ gacc,
                            float* __restrict__ out_depth) {
  int ray = blockIdx.x * blockDim.x + threadIdx.x;
  if (ray >= N_RAYS) return;
  float o[3], d[3];
  float tmin = ray_tmin(ro, rd, ray, o, d);
  float T = 1.0f, acc = 0.0f, depth = 0.0f;
  size_t base = (size_t)ray * N_SAMP;
  for (int s = 0; s < N_SAMP; ++s) {
    float sg = gsigma[base + s];
    float dist = (s < N_SAMP - 1) ? STEP_F : 0.0f;
    float alpha = 1.0f - __expf(-sg * dist * DIST_SCALE);
    float w = alpha * T;
    gweight[base + s] = w;
    acc += w;
    depth += w * (tmin + STEP_F * (float)s);
    T *= (1.0f - alpha + 1e-10f);
  }
  out_depth[ray] = depth;
  gacc[ray] = acc;
}

// --------------------------- kernel 3: WMMA MLP ----------------------------
// 256 threads = 8 waves, 128 points/block (one ray per block).
// Weights staged global->LDS with the Tensor Data Mover, double-buffered so
// the W2 DMA overlaps layer 1 and the W3 DMA overlaps layer 2.
__global__ void __launch_bounds__(256)
k_mlp(const float* __restrict__ rd, const float* __restrict__ gfeat,
      const float* __restrict__ gweight,
      const unsigned short* __restrict__ W1B, const unsigned short* __restrict__ W2B,
      const unsigned short* __restrict__ W3B,
      const float* __restrict__ b1, const float* __restrict__ b2,
      const float* __restrict__ b3,
      float* __restrict__ partials) {
  __shared__ __align__(16) unsigned short lds_A[TILE_M * KPAD];   // 40 KB act tile
  __shared__ __align__(16) unsigned short lds_Wa[5 * 128 * 32];   // 40 KB (W1 / W3)
  __shared__ __align__(16) unsigned short lds_Wb[4 * 128 * 32];   // 32 KB (W2)
  __shared__ float lds_red[8][2][3];

  int blk  = blockIdx.x;
  int t    = threadIdx.x;
  int lane = t & 31;
  int wave = t >> 5;
  int p0   = blk * TILE_M;
  int ray  = p0 >> 9;               // 128 | 512 -> whole block is one ray
  float v0 = rd[ray * 3 + 0], v1 = rd[ray * 3 + 1], v2 = rd[ray * 3 + 2];

  __builtin_prefetch(W1B, 0, 0);    // global_prefetch_b8 into L2 for weights
  __builtin_prefetch(W2B, 0, 0);

  // kick off W1 staging, overlap with mlp_in construction
  stage_weights(lds_Wa, W1B, W1_DW, t);

  // ---- build mlp_in (bf16) rows: [feat27 | view3 | sin54 | cos54 | sin6 | cos6 | 0-pad]
  {
    int m = t >> 1, half = t & 1;
    const float* fr = gfeat + (size_t)(p0 + m) * APP_DIM;
    for (int e = half * 80; e < half * 80 + 80; ++e) {
      float val;
      if (e < 27)       val = fr[e];
      else if (e < 30)  val = (e == 27) ? v0 : (e == 28) ? v1 : v2;
      else if (e < 84)  { int j = e - 30;  float x = fr[j >> 1] * ((j & 1) ? 2.0f : 1.0f); val = __sinf(x); }
      else if (e < 138) { int j = e - 84;  float x = fr[j >> 1] * ((j & 1) ? 2.0f : 1.0f); val = __cosf(x); }
      else if (e < 144) { int j = e - 138; float b = ((j >> 1) == 0) ? v0 : ((j >> 1) == 1) ? v1 : v2;
                          val = __sinf(b * ((j & 1) ? 2.0f : 1.0f)); }
      else if (e < 150) { int j = e - 144; float b = ((j >> 1) == 0) ? v0 : ((j >> 1) == 1) ? v1 : v2;
                          val = __cosf(b * ((j & 1) ? 2.0f : 1.0f)); }
      else val = 0.0f;
      lds_A[m * KPAD + e] = f2bf(val);
    }
  }

  // kick off W2 staging (independent LDS buffer), then make sure W1 landed
  stage_weights(lds_Wb, W2B, W2_DW, t);
  wait_tensor<1>(t);                // W1 complete (TDM ops in-order per wave)
  __syncthreads();

  int mrow  = wave * 16 + (lane & 15);  // this lane's A row
  int khalf = lane >> 4;                // K-half select per ISA A layout
  v8f acc[8];

  // ------------------- layer 1: [16 x 160] @ [160 x 128] -------------------
#pragma unroll
  for (int nt = 0; nt < 8; ++nt) {
    float b = b1[nt * 16 + (lane & 15)];
#pragma unroll
    for (int i = 0; i < 8; ++i) acc[nt][i] = b;
  }
#pragma unroll
  for (int kc = 0; kc < 5; ++kc) {
    Frag A;
    const u32x4* ar = (const u32x4*)&lds_A[mrow * KPAD + kc * 32];
    A.q[0] = ar[khalf];
    A.q[1] = ar[2 + khalf];
#pragma unroll
    for (int nt = 0; nt < 8; ++nt) {
      Frag B;
      const u32x4* br =
        (const u32x4*)&lds_Wa[(kc * 128 + nt * 16 + (lane & 15)) * 32];
      B.q[0] = br[khalf];
      B.q[1] = br[2 + khalf];
      acc[nt] = __builtin_amdgcn_wmma_f32_16x16x32_bf16(
                  false, A.v, false, B.v, (short)0, acc[nt], false, false);
    }
  }
  // relu + writeback (rows are wave-exclusive; cols 0..127 overwrite mlp_in)
#pragma unroll
  for (int nt = 0; nt < 8; ++nt) {
    int n = nt * 16 + (lane & 15);
#pragma unroll
    for (int r = 0; r < 8; ++r) {
      int m = wave * 16 + r + khalf * 8;
      lds_A[m * KPAD + n] = f2bf(fmaxf(acc[nt][r], 0.0f));
    }
  }
  __syncthreads();   // all waves done reading lds_Wa (W1)

  // stage W3 into lds_Wa (overlaps layer 2 on the TDM path); ensure W2 landed
  stage_weights(lds_Wa, W3B, W3_DW, t);
  wait_tensor<1>(t);                // W2 complete
  __syncthreads();

  // ------------------- layer 2: [16 x 128] @ [128 x 128] -------------------
#pragma unroll
  for (int nt = 0; nt < 8; ++nt) {
    float b = b2[nt * 16 + (lane & 15)];
#pragma unroll
    for (int i = 0; i < 8; ++i) acc[nt][i] = b;
  }
#pragma unroll
  for (int kc = 0; kc < 4; ++kc) {
    Frag A;
    const u32x4* ar = (const u32x4*)&lds_A[mrow * KPAD + kc * 32];
    A.q[0] = ar[khalf];
    A.q[1] = ar[2 + khalf];
#pragma unroll
    for (int nt = 0; nt < 8; ++nt) {
      Frag B;
      const u32x4* br =
        (const u32x4*)&lds_Wb[(kc * 128 + nt * 16 + (lane & 15)) * 32];
      B.q[0] = br[khalf];
      B.q[1] = br[2 + khalf];
      acc[nt] = __builtin_amdgcn_wmma_f32_16x16x32_bf16(
                  false, A.v, false, B.v, (short)0, acc[nt], false, false);
    }
  }
#pragma unroll
  for (int nt = 0; nt < 8; ++nt) {
    int n = nt * 16 + (lane & 15);
#pragma unroll
    for (int r = 0; r < 8; ++r) {
      int m = wave * 16 + r + khalf * 8;
      lds_A[m * KPAD + n] = f2bf(fmaxf(acc[nt][r], 0.0f));
    }
  }
  wait_tensor<0>(t);                // W3 complete
  __syncthreads();

  // ------------------- layer 3: [16 x 128] @ [128 x 16(3)] -----------------
  v8f acc3;
  {
    float b = ((lane & 15) < 3) ? b3[lane & 15] : 0.0f;
#pragma unroll
    for (int i = 0; i < 8; ++i) acc3[i] = b;
  }
#pragma unroll
  for (int kc = 0; kc < 4; ++kc) {
    Frag A;
    const u32x4* ar = (const u32x4*)&lds_A[mrow * KPAD + kc * 32];
    A.q[0] = ar[khalf];
    A.q[1] = ar[2 + khalf];
    Frag B;
    const u32x4* br =
      (const u32x4*)&lds_Wa[(kc * 16 + (lane & 15)) * 32];
    B.q[0] = br[khalf];
    B.q[1] = br[2 + khalf];
    acc3 = __builtin_amdgcn_wmma_f32_16x16x32_bf16(
             false, A.v, false, B.v, (short)0, acc3, false, false);
  }

  // sigmoid + weight mask + deterministic block reduction
  int n = lane & 15;
  if (n < 3) {
    float partial = 0.0f;
#pragma unroll
    for (int r = 0; r < 8; ++r) {
      int m = wave * 16 + r + khalf * 8;
      float w = gweight[(size_t)(p0 + m)];
      if (w > W_THRES) {
        float rgb = 1.0f / (1.0f + __expf(-acc3[r]));
        partial += w * rgb;
      }
    }
    lds_red[wave][khalf][n] = partial;
  }
  __syncthreads();
  if (t < 3) {
    float s = 0.0f;
#pragma unroll
    for (int w = 0; w < 8; ++w)
#pragma unroll
      for (int h = 0; h < 2; ++h) s += lds_red[w][h][t];
    partials[blk * 3 + t] = s;
  }
}

// --------------------------- kernel 4: finalize ----------------------------
__global__ void k_final(const float* __restrict__ partials,
                        const float* __restrict__ gacc,
                        float* __restrict__ out_rgb) {
  int i = blockIdx.x * blockDim.x + threadIdx.x;
  if (i >= N_RAYS * 3) return;
  int ray = i / 3, c = i % 3;
  float s = 0.0f;
#pragma unroll
  for (int tq = 0; tq < 4; ++tq) s += partials[(ray * 4 + tq) * 3 + c];
  float v = s + (1.0f - gacc[ray]);
  out_rgb[i] = fminf(fmaxf(v, 0.0f), 1.0f);
}

// ---------------------------------------------------------------------------
extern "C" void kernel_launch(void* const* d_in, const int* in_sizes, int n_in,
                              void* d_out, int out_size, void* d_ws, size_t ws_size,
                              hipStream_t stream) {
  const float* ro    = (const float*)d_in[0];
  const float* rdir  = (const float*)d_in[1];
  const float* dp0   = (const float*)d_in[2];
  const float* dl0   = (const float*)d_in[3];
  const float* ap0   = (const float*)d_in[4];
  const float* al0   = (const float*)d_in[5];
  const float* dp1   = (const float*)d_in[6];
  const float* dl1   = (const float*)d_in[7];
  const float* ap1   = (const float*)d_in[8];
  const float* al1   = (const float*)d_in[9];
  const float* dp2   = (const float*)d_in[10];
  const float* dl2   = (const float*)d_in[11];
  const float* ap2   = (const float*)d_in[12];
  const float* al2   = (const float*)d_in[13];
  const float* basis = (const float*)d_in[14];
  const float* w1    = (const float*)d_in[15];
  const float* b1    = (const float*)d_in[16];
  const float* w2    = (const float*)d_in[17];
  const float* b2    = (const float*)d_in[18];
  const float* w3    = (const float*)d_in[19];
  const float* b3    = (const float*)d_in[20];
  float* out = (float*)d_out;

  // workspace layout (bytes)
  char* ws = (char*)d_ws;
  unsigned short* W1B = (unsigned short*)(ws + 0);        // 40960
  unsigned short* W2B = (unsigned short*)(ws + 40960);    // 32768
  unsigned short* W3B = (unsigned short*)(ws + 73728);    // 4096
  float* partials     = (float*)(ws + 77824);             // 4096*3*4 = 49152
  float* gacc         = (float*)(ws + 126976);            // 1024*4
  float* gsigma       = (float*)(ws + 131072);            // NPTS*4
  float* gweight      = (float*)(ws + 2228224);           // NPTS*4
  float* gfeat        = (float*)(ws + 4325376);           // NPTS*27*4 (~56.6 MB)

  k_prep<<<152, 256, 0, stream>>>(w1, w2, w3, W1B, W2B, W3B);
  k_points<<<NPTS / 256, 256, 0, stream>>>(ro, rdir,
      dp0, dl0, ap0, al0, dp1, dl1, ap1, al1, dp2, dl2, ap2, al2,
      basis, gsigma, gfeat);
  k_composite<<<N_RAYS / 256, 256, 0, stream>>>(ro, rdir, gsigma, gweight, gacc,
                                                out + N_RAYS * 3);
  k_mlp<<<NPTS / TILE_M, 256, 0, stream>>>(rdir, gfeat, gweight,
                                           W1B, W2B, W3B, b1, b2, b3, partials);
  k_final<<<12, 256, 0, stream>>>(partials, gacc, out);
}
